// DSEN_34351148434016
// MI455X (gfx1250) — matrix operand order repaired and checked by reading, not compiled, not run on hardware
//
#include <hip/hip_runtime.h>

// ---------------------------------------------------------------------------
// MI455X (gfx1250) implementation. wave32. Matrix core via
// v_wmma_f32_16x16x32_f16. EdgeConv MLPs (≈173 GFLOP) are the hot path.
// ---------------------------------------------------------------------------

typedef _Float16 half_t;
typedef __attribute__((ext_vector_type(16))) _Float16 v16h;
typedef __attribute__((ext_vector_type(8)))  float    v8f;

#define NGRAPH 128
#define NCH    30
#define TLEN   3600
#define SEGL   200
#define NSEG   18
#define BNEPS  1e-5f

// ---- WMMA fragment builders (layouts per CDNA5 ISA 7.12.2, wave32) --------

// A-matrix 16x32 f16: lane<16 -> row (lane&15), K {0..7,16..23};
// lane>=16 -> same row set, K {8..15,24..31}. p points at (row, kb).
__device__ __forceinline__ v16h fragA_from(const half_t* p) {
  v16h r;
#pragma unroll
  for (int v = 0; v < 8; ++v) {
    const int ko = 2 * (v & 3) + ((v & 4) ? 16 : 0);
    r[2 * v]     = p[ko];
    r[2 * v + 1] = p[ko + 1];
  }
  return r;
}

// B-matrix 32x16 f16 built from W^T (N,K) row-major: lane<16 -> col (lane&15),
// K {0..7,16..23}; lane>=16 -> K {8..15,24..31}.  p points at (col, kb).
__device__ __forceinline__ v16h fragB_from(const half_t* p) {
  v16h r;
#pragma unroll
  for (int v = 0; v < 16; ++v) r[v] = p[(v & 8) ? 16 + (v & 7) : v];
  return r;
}

__device__ __forceinline__ v8f wmma16x16x32(v16h a, v16h b, v8f c) {
  return __builtin_amdgcn_wmma_f32_16x16x32_f16(false, a, false, b,
                                                (short)0, c, false, false);
}

// ---- Stage 1: depthwise conv k=32 pad=15 + BN + ELU + adaptive pool 199->100
__global__ __launch_bounds__(128) void stage1_kernel(
    const float* __restrict__ x, const float* __restrict__ w1,
    const float* __restrict__ g, const float* __restrict__ b,
    const float* __restrict__ m, const float* __restrict__ v,
    float* __restrict__ out) {
  __shared__ float sIn[SEGL];
  __shared__ float sW[32];
  const int blk = blockIdx.x;          // 0 .. 2304*30-1
  const int c   = blk % NCH;
  const int seg = blk / NCH;           // b*18+g
  const int bb  = seg / NSEG, gg = seg % NSEG;
  const float* inp = x + (size_t)(bb * NCH + c) * TLEN + gg * SEGL;
  const int tid = threadIdx.x;
  for (int i = tid; i < SEGL; i += 128) sIn[i] = inp[i];
  if (tid < 32) sW[tid] = w1[c * 32 + tid];
  __syncthreads();
  const float scale = rsqrtf(v[c] + BNEPS) * g[c];
  const float shift = b[c] - m[c] * scale;
  if (tid < 100) {
    const int start = (tid * 199) / 100;
    const int end   = ((tid + 1) * 199 + 99) / 100;  // ceil
    float s = 0.f;
    for (int t = start; t < end; ++t) {
      float acc = 0.f;
#pragma unroll
      for (int k = 0; k < 32; ++k) {
        const int ii = t - 15 + k;
        const float xv = (ii >= 0 && ii < SEGL) ? sIn[ii] : 0.f;
        acc = fmaf(xv, sW[k], acc);
      }
      acc = acc * scale + shift;
      acc = acc > 0.f ? acc : (expf(acc) - 1.f);    // ELU
      s += acc;
    }
    out[(size_t)(bb * NCH + c) * 1800 + gg * 100 + tid] = s / (float)(end - start);
  }
}

// ---- Stage 2: depthwise conv k=100 pad=0 + BN + ELU + adaptive pool 1701->128
__global__ __launch_bounds__(256) void stage2_kernel(
    const float* __restrict__ h, const float* __restrict__ w2,
    const float* __restrict__ g, const float* __restrict__ b,
    const float* __restrict__ m, const float* __restrict__ v,
    float* __restrict__ nodes) {
  __shared__ float sIn[1800];
  __shared__ float sW[100];
  __shared__ float sOut[1701];
  const int bc  = blockIdx.x;          // b*30+c
  const int c   = bc % NCH;
  const int tid = threadIdx.x;
  for (int i = tid; i < 1800; i += 256) sIn[i] = h[(size_t)bc * 1800 + i];
  for (int i = tid; i < 100; i += 256)  sW[i]  = w2[c * 100 + i];
  __syncthreads();
  const float scale = rsqrtf(v[c] + BNEPS) * g[c];
  const float shift = b[c] - m[c] * scale;
  for (int t = tid; t < 1701; t += 256) {
    float acc = 0.f;
    for (int k = 0; k < 100; ++k) acc = fmaf(sIn[t + k], sW[k], acc);
    acc = acc * scale + shift;
    sOut[t] = acc > 0.f ? acc : (expf(acc) - 1.f);
  }
  __syncthreads();
  if (tid < 128) {
    const int start = (tid * 1701) / 128;
    const int end   = ((tid + 1) * 1701 + 127) / 128;
    float s = 0.f;
    for (int t = start; t < end; ++t) s += sOut[t];
    nodes[(size_t)bc * 128 + tid] = s / (float)(end - start);
  }
}

// ---- weight transpose + f16 convert:  W(K,O) -> Wt(O,K) f16 ---------------
__global__ void transpose_f16_kernel(const float* __restrict__ W,
                                     half_t* __restrict__ Wt, int K, int O) {
  const int idx = blockIdx.x * blockDim.x + threadIdx.x;
  if (idx < K * O) {
    const int k = idx / O, o = idx % O;
    Wt[o * K + k] = (half_t)W[idx];
  }
}

// ---- EdgeConv (fused gather + MLP(2 GEMMs) + ReLU + BN + per-dst max) -----
// One workgroup (4 waves) per (graph, dst). 32 edge rows (29 real + 3 dup).
template <int F, int O>
__global__ __launch_bounds__(128) void edgeconv_kernel(
    const float*  __restrict__ X,     // (G*30, F) node features
    const half_t* __restrict__ W1t,   // (O, 2F) f16, transposed
    const float*  __restrict__ b1,
    const half_t* __restrict__ W2t,   // (O, O) f16, transposed
    const float*  __restrict__ b2,
    const float*  __restrict__ bng, const float* __restrict__ bnb,
    const float*  __restrict__ bnm, const float* __restrict__ bnv,
    float* __restrict__ Y) {          // (G*30, O)
  static_assert(F % 32 == 0 && O % 16 == 0, "tile shape");
  constexpr int K1 = 2 * F;
  __shared__ half_t sXd[F];           // x_dst (shared across all 32 rows)
  __shared__ half_t sDiff[32 * F];    // x_src - x_dst per row
  __shared__ half_t sH1[32 * O];      // hidden activations
  const int gid = blockIdx.x / NCH;
  const int dst = blockIdx.x % NCH;
  const int nodeBase = gid * NCH;
  const int tid = threadIdx.x;

  for (int k = tid; k < F; k += 128)
    sXd[k] = (half_t)X[(size_t)(nodeBase + dst) * F + k];
  for (int idx = tid; idx < 32 * F; idx += 128) {
    const int r = idx / F, k = idx % F;
    const int jj = (r < 29) ? r : 0;               // pad rows duplicate edge 0
    const int sn = jj + (jj >= dst ? 1 : 0);       // skip self-loop
    sDiff[idx] = (half_t)(X[(size_t)(nodeBase + sn) * F + k] -
                          X[(size_t)(nodeBase + dst) * F + k]);
  }
  __syncthreads();

  const int lane = tid & 31, wave = tid >> 5;
  const int mrow = lane & 15;
  const int aoff = (lane & 16) ? 8 : 0;   // A kb offset
  const int boff = (lane & 16) ? 8 : 0;   // B kb offset

  // ---- GEMM1: (32 x 2F) x (2F x O), + b1, ReLU -> sH1 (f16)
  for (int nt = wave; nt < O / 16; nt += 4) {
    v8f acc0 = {}, acc1 = {};
    for (int kt = 0; kt < K1 / 32; ++kt) {
      const v16h bf = fragB_from(W1t + (size_t)(nt * 16 + mrow) * K1 + kt * 32 + boff);
      v16h a0, a1;
      if (kt * 32 < F) {                 // [x_i] half: identical rows
        a0 = fragA_from(sXd + kt * 32 + aoff);
        a1 = a0;
      } else {                           // [x_j - x_i] half
        const int kb = kt * 32 - F + aoff;
        a0 = fragA_from(sDiff + mrow * F + kb);
        a1 = fragA_from(sDiff + (16 + mrow) * F + kb);
      }
      acc0 = wmma16x16x32(a0, bf, acc0);
      acc1 = wmma16x16x32(a1, bf, acc1);
    }
    const int c = nt * 16 + mrow;
    const float bias = b1[c];
#pragma unroll
    for (int r = 0; r < 8; ++r) {
      sH1[(aoff + r) * O + c]      = (half_t)fmaxf(acc0[r] + bias, 0.f);
      sH1[(16 + aoff + r) * O + c] = (half_t)fmaxf(acc1[r] + bias, 0.f);
    }
  }
  __syncthreads();

  // ---- GEMM2: (32 x O) x (O x O), + b2, ReLU, BN, max over 32 rows -> Y row
  for (int nt = wave; nt < O / 16; nt += 4) {
    v8f acc0 = {}, acc1 = {};
    for (int kt = 0; kt < O / 32; ++kt) {
      const v16h bf = fragB_from(W2t + (size_t)(nt * 16 + mrow) * O + kt * 32 + boff);
      const v16h a0 = fragA_from(sH1 + mrow * O + kt * 32 + aoff);
      const v16h a1 = fragA_from(sH1 + (16 + mrow) * O + kt * 32 + aoff);
      acc0 = wmma16x16x32(a0, bf, acc0);
      acc1 = wmma16x16x32(a1, bf, acc1);
    }
    const int c = nt * 16 + mrow;
    const float bias  = b2[c];
    const float scale = bng[c] * rsqrtf(bnv[c] + BNEPS);
    const float shift = bnb[c] - bnm[c] * scale;
    float mx = -__builtin_inff();
#pragma unroll
    for (int r = 0; r < 8; ++r) {
      const float v0 = fmaxf(acc0[r] + bias, 0.f) * scale + shift;
      const float v1 = fmaxf(acc1[r] + bias, 0.f) * scale + shift;
      mx = fmaxf(mx, fmaxf(v0, v1));
    }
    mx = fmaxf(mx, __shfl_xor(mx, 16, 32));  // lanes n / n+16 share column c
    if (lane < 16) Y[(size_t)(nodeBase + dst) * O + c] = mx;
  }
}

// ---- graph-level max pool over the 30 nodes of each graph -----------------
__global__ __launch_bounds__(256) void graph_pool_kernel(
    const float* __restrict__ Y, float* __restrict__ P, int O) {
  const int g = blockIdx.x;
  for (int c = threadIdx.x; c < O; c += blockDim.x) {
    float mx = -__builtin_inff();
    for (int n = 0; n < NCH; ++n)
      mx = fmaxf(mx, Y[(size_t)(g * NCH + n) * O + c]);
    P[(size_t)g * O + c] = mx;
  }
}

// ---- concat [p1|p2|p3] -> f16 (128, 896) ----------------------------------
__global__ void build_concat_kernel(const float* __restrict__ p1,
                                    const float* __restrict__ p2,
                                    const float* __restrict__ p3,
                                    half_t* __restrict__ out) {
  const int idx = blockIdx.x * blockDim.x + threadIdx.x;
  if (idx < NGRAPH * 896) {
    const int r = idx / 896, c = idx % 896;
    float v;
    if (c < 128)      v = p1[r * 128 + c];
    else if (c < 384) v = p2[r * 256 + (c - 128)];
    else              v = p3[r * 512 + (c - 384)];
    out[idx] = (half_t)v;
  }
}

// ---- generic WMMA GEMM + bias + ReLU for FC head --------------------------
// A (M,K) f16 row-major; Bt (N,K) f16; one wave per 16x16 output tile.
template <bool HALF_OUT>
__global__ __launch_bounds__(128) void fc_wmma_kernel(
    const half_t* __restrict__ A, const half_t* __restrict__ Bt,
    const float* __restrict__ bias, void* __restrict__ Out,
    int M, int K, int N) {
  const int tiles = (M / 16) * (N / 16);
  const int wid = blockIdx.x * 4 + (threadIdx.x >> 5);
  if (wid >= tiles) return;
  const int mt = wid / (N / 16), nt = wid % (N / 16);
  const int lane = threadIdx.x & 31;
  const int mrow = lane & 15;
  const int aoff = (lane & 16) ? 8 : 0;
  const int boff = (lane & 16) ? 8 : 0;
  v8f acc = {};
  for (int kt = 0; kt < K / 32; ++kt) {
    const v16h a = fragA_from(A + (size_t)(mt * 16 + mrow) * K + kt * 32 + aoff);
    const v16h b = fragB_from(Bt + (size_t)(nt * 16 + mrow) * K + kt * 32 + boff);
    acc = wmma16x16x32(a, b, acc);
  }
  const int c = nt * 16 + mrow;
  const float bb = bias[c];
  const int rbase = mt * 16 + aoff;
#pragma unroll
  for (int r = 0; r < 8; ++r) {
    const float v = fmaxf(acc[r] + bb, 0.f);
    if (HALF_OUT) ((half_t*)Out)[(size_t)(rbase + r) * N + c] = (half_t)v;
    else          ((float*)Out)[(size_t)(rbase + r) * N + c] = v;
  }
}

// ---------------------------------------------------------------------------
extern "C" void kernel_launch(void* const* d_in, const int* in_sizes, int n_in,
                              void* d_out, int out_size, void* d_ws, size_t ws_size,
                              hipStream_t stream) {
  // setup_inputs() insertion order
  const float* x     = (const float*)d_in[0];
  const float* w1    = (const float*)d_in[1];
  const float* bn1g  = (const float*)d_in[2];
  const float* bn1b  = (const float*)d_in[3];
  const float* bn1m  = (const float*)d_in[4];
  const float* bn1v  = (const float*)d_in[5];
  const float* w2    = (const float*)d_in[6];
  const float* bn2g  = (const float*)d_in[7];
  const float* bn2b  = (const float*)d_in[8];
  const float* bn2m  = (const float*)d_in[9];
  const float* bn2v  = (const float*)d_in[10];
  const float* c1w1  = (const float*)d_in[11];
  const float* c1b1  = (const float*)d_in[12];
  const float* c1w2  = (const float*)d_in[13];
  const float* c1b2  = (const float*)d_in[14];
  const float* c1bng = (const float*)d_in[15];
  const float* c1bnb = (const float*)d_in[16];
  const float* c1bnm = (const float*)d_in[17];
  const float* c1bnv = (const float*)d_in[18];
  const float* c2w1  = (const float*)d_in[19];
  const float* c2b1  = (const float*)d_in[20];
  const float* c2w2  = (const float*)d_in[21];
  const float* c2b2  = (const float*)d_in[22];
  const float* c2bng = (const float*)d_in[23];
  const float* c2bnb = (const float*)d_in[24];
  const float* c2bnm = (const float*)d_in[25];
  const float* c2bnv = (const float*)d_in[26];
  const float* c3w1  = (const float*)d_in[27];
  const float* c3b1  = (const float*)d_in[28];
  const float* c3w2  = (const float*)d_in[29];
  const float* c3b2  = (const float*)d_in[30];
  const float* c3bng = (const float*)d_in[31];
  const float* c3bnb = (const float*)d_in[32];
  const float* c3bnm = (const float*)d_in[33];
  const float* c3bnv = (const float*)d_in[34];
  const float* l1w   = (const float*)d_in[35];
  const float* l1b   = (const float*)d_in[36];
  const float* l2w   = (const float*)d_in[37];
  const float* l2b   = (const float*)d_in[38];
  // d_in[39]=edge_index, d_in[40]=batch: structure is known statically.

  // workspace carve-up
  char* base = (char*)d_ws;
  size_t off = 0;
  auto carve = [&](size_t bytes) -> char* {
    char* p = base + off;
    off = (off + bytes + 255) & ~(size_t)255;
    return p;
  };
  float*  h1       = (float*)carve((size_t)NGRAPH * NCH * 1800 * 4);
  float*  nodes0   = (float*)carve((size_t)NGRAPH * NCH * 128 * 4);
  float*  x1       = (float*)carve((size_t)NGRAPH * NCH * 128 * 4);
  float*  x2       = (float*)carve((size_t)NGRAPH * NCH * 256 * 4);
  float*  x3       = (float*)carve((size_t)NGRAPH * NCH * 512 * 4);
  float*  p1       = (float*)carve((size_t)NGRAPH * 128 * 4);
  float*  p2       = (float*)carve((size_t)NGRAPH * 256 * 4);
  float*  p3       = (float*)carve((size_t)NGRAPH * 512 * 4);
  half_t* concat16 = (half_t*)carve((size_t)NGRAPH * 896 * 2);
  half_t* fc1h     = (half_t*)carve((size_t)NGRAPH * 256 * 2);
  half_t* c1W1t    = (half_t*)carve((size_t)256 * 128 * 2);
  half_t* c1W2t    = (half_t*)carve((size_t)128 * 128 * 2);
  half_t* c2W1t    = (half_t*)carve((size_t)256 * 256 * 2);
  half_t* c2W2t    = (half_t*)carve((size_t)256 * 256 * 2);
  half_t* c3W1t    = (half_t*)carve((size_t)512 * 512 * 2);
  half_t* c3W2t    = (half_t*)carve((size_t)512 * 512 * 2);
  half_t* l1t      = (half_t*)carve((size_t)896 * 256 * 2);
  half_t* l2t      = (half_t*)carve((size_t)256 * 128 * 2);
  (void)ws_size; (void)in_sizes; (void)n_in; (void)out_size;

  // weight prep (f16 transpose)
  auto tpose = [&](const float* W, half_t* Wt, int K, int O) {
    transpose_f16_kernel<<<(K * O + 255) / 256, 256, 0, stream>>>(W, Wt, K, O);
  };
  tpose(c1w1, c1W1t, 256, 128);
  tpose(c1w2, c1W2t, 128, 128);
  tpose(c2w1, c2W1t, 256, 256);
  tpose(c2w2, c2W2t, 256, 256);
  tpose(c3w1, c3W1t, 512, 512);
  tpose(c3w2, c3W2t, 512, 512);
  tpose(l1w, l1t, 896, 256);
  tpose(l2w, l2t, 256, 128);

  // stage 1: (2304 segs x 30 ch)
  stage1_kernel<<<NGRAPH * NSEG * NCH, 128, 0, stream>>>(
      x, w1, bn1g, bn1b, bn1m, bn1v, h1);
  // stage 2: (3840 rows)
  stage2_kernel<<<NGRAPH * NCH, 256, 0, stream>>>(
      h1, w2, bn2g, bn2b, bn2m, bn2v, nodes0);

  // EdgeConv layers: one block per (graph, dst) = 3840 blocks
  edgeconv_kernel<128, 128><<<NGRAPH * NCH, 128, 0, stream>>>(
      nodes0, c1W1t, c1b1, c1W2t, c1b2, c1bng, c1bnb, c1bnm, c1bnv, x1);
  graph_pool_kernel<<<NGRAPH, 256, 0, stream>>>(x1, p1, 128);
  edgeconv_kernel<128, 256><<<NGRAPH * NCH, 128, 0, stream>>>(
      x1, c2W1t, c2b1, c2W2t, c2b2, c2bng, c2bnb, c2bnm, c2bnv, x2);
  graph_pool_kernel<<<NGRAPH, 256, 0, stream>>>(x2, p2, 256);
  edgeconv_kernel<256, 512><<<NGRAPH * NCH, 128, 0, stream>>>(
      x2, c3W1t, c3b1, c3W2t, c3b2, c3bng, c3bnb, c3bnm, c3bnv, x3);
  graph_pool_kernel<<<NGRAPH, 256, 0, stream>>>(x3, p3, 512);

  // FC head
  build_concat_kernel<<<(NGRAPH * 896 + 255) / 256, 256, 0, stream>>>(
      p1, p2, p3, concat16);
  // (128,896)x(896,256): 8*16=128 tiles -> 32 blocks of 4 waves
  fc_wmma_kernel<true><<<32, 128, 0, stream>>>(
      concat16, l1t, l1b, (void*)fc1h, 128, 896, 256);
  // (128,256)x(256,128): 8*8=64 tiles -> 16 blocks
  fc_wmma_kernel<false><<<16, 128, 0, stream>>>(
      fc1h, l2t, l2b, d_out, 128, 256, 128);
}
